// Mamba_adapter_83279415869727
// MI455X (gfx1250) — matrix operand order repaired and checked by reading, not compiled, not run on hardware
//
#include <hip/hip_runtime.h>
#include <math.h>

// Problem constants (match reference)
#define BSZ   16
#define LTOT  2048          // concat of two L=1024 halves
#define DIMX  128
#define DI    256           // D_INNER
#define DS    16            // D_STATE
#define RK    8             // DT_RANK
#define NXP   48            // DT_RANK + 2*D_STATE = 40, padded to 48 for WMMA tiles
#define NTOK  (BSZ*LTOT)    // 32768 tokens
#define TCH   64            // scan timestep chunk staged per TDM transfer

typedef _Float16 v8h  __attribute__((ext_vector_type(8)));
typedef _Float16 v16h __attribute__((ext_vector_type(16)));
typedef float    v8f  __attribute__((ext_vector_type(8)));
typedef unsigned int u32x4 __attribute__((ext_vector_type(4)));
typedef unsigned int u32x8 __attribute__((ext_vector_type(8)));

// ---------------------------------------------------------------------------
// WMMA fragment loaders per CDNA5 ISA 7.12.2 (wave32).
// A (16x32 f16): lanes 0-15 = rows M, hold K=0..7 / K=16..23; lanes 16-31 the rest.
// B (32x16 f16): lane&15 = column N, lane>>4 selects K half; element j <-> K=j+8*half (+8 if j>=8).
// C/D (16x16 f32): VGPR r -> row r + 8*(lane>>4), col = lane&15.
// ---------------------------------------------------------------------------
__device__ __forceinline__ v16h load_a_frag(const _Float16* __restrict__ rowbase,
                                            int stride, int m, int half, int kbase) {
  const _Float16* p = rowbase + (size_t)m * stride + kbase + 8 * half;
  v8h lo = *(const v8h*)(p);
  v8h hi = *(const v8h*)(p + 16);
  v16h a;
#pragma unroll
  for (int j = 0; j < 8; ++j) { a[j] = lo[j]; a[j + 8] = hi[j]; }
  return a;
}

__device__ __forceinline__ v16h load_b_frag(const _Float16* __restrict__ w,
                                            int K, int row, int half, int kbase) {
  const _Float16* p = w + (size_t)row * K + kbase + 8 * half;
  v8h lo = *(const v8h*)(p);
  v8h hi = *(const v8h*)(p + 16);
  v16h b;
#pragma unroll
  for (int j = 0; j < 8; ++j) { b[j] = lo[j]; b[j + 8] = hi[j]; }
  return b;
}

__device__ __forceinline__ v8f wmma_f16(v16h a, v16h b, v8f c) {
  return __builtin_amdgcn_wmma_f32_16x16x32_f16(false, a, false, b, (short)0, c,
                                                false, false);
}

// ---------------------------------------------------------------------------
// 0) concat x = [x_adap ; xi_adap] along L
// ---------------------------------------------------------------------------
__global__ void k_concat(const float* __restrict__ xa, const float* __restrict__ xi,
                         float* __restrict__ x) {
  int idx = blockIdx.x * blockDim.x + threadIdx.x;   // NTOK*DIMX
  int tok = idx >> 7, c = idx & 127;
  int b = tok >> 11, t = tok & 2047;
  x[idx] = (t < 1024) ? xa[(((size_t)(b << 10)) + t) * DIMX + c]
                      : xi[(((size_t)(b << 10)) + (t - 1024)) * DIMX + c];
}

// ---------------------------------------------------------------------------
// 1) per-layer weight conversion to f16 (Win, Wx padded 40->48, Wout)
// ---------------------------------------------------------------------------
__global__ void k_wconv(const float* __restrict__ Win, const float* __restrict__ Wx,
                        const float* __restrict__ Wout, _Float16* __restrict__ Win16,
                        _Float16* __restrict__ Wx16, _Float16* __restrict__ Wout16) {
  int idx = blockIdx.x * blockDim.x + threadIdx.x;
  if (idx < 512 * DIMX) {
    Win16[idx] = (_Float16)Win[idx];
  } else if (idx < 512 * DIMX + NXP * DI) {
    int j = idx - 512 * DIMX;
    int e = j / DI, d = j % DI;
    Wx16[j] = (e < (RK + 2 * DS)) ? (_Float16)Wx[e * DI + d] : (_Float16)0.f;
  } else if (idx < 512 * DIMX + NXP * DI + DIMX * DI) {
    int j = idx - 512 * DIMX - NXP * DI;
    Wout16[j] = (_Float16)Wout[j];
  }
}

// ---------------------------------------------------------------------------
// 2) LayerNorm (+ optional time reversal) -> f16 activations, wave per token
// ---------------------------------------------------------------------------
__global__ void k_ln(const float* __restrict__ x, const float* __restrict__ w,
                     const float* __restrict__ bb, _Float16* __restrict__ h, int rev) {
  int tok = blockIdx.x * 8 + (threadIdx.x >> 5);
  int lane = threadIdx.x & 31;
  int b = tok >> 11, t = tok & 2047;
  int ts = rev ? (2047 - t) : t;
  float4 v = *(const float4*)(x + (((size_t)(b << 11) + ts) << 7) + lane * 4);
  float s = v.x + v.y + v.z + v.w;
#pragma unroll
  for (int o = 16; o > 0; o >>= 1) s += __shfl_xor(s, o, 32);
  float mu = s * (1.f / 128.f);
  float dx = v.x - mu, dy = v.y - mu, dz = v.z - mu, dw = v.w - mu;
  float q = dx * dx + dy * dy + dz * dz + dw * dw;
#pragma unroll
  for (int o = 16; o > 0; o >>= 1) q += __shfl_xor(q, o, 32);
  float rs = rsqrtf(q * (1.f / 128.f) + 1e-5f);
  int c = lane * 4;
  _Float16* hp = h + (((size_t)tok) << 7) + c;
  hp[0] = (_Float16)(dx * rs * w[c + 0] + bb[c + 0]);
  hp[1] = (_Float16)(dy * rs * w[c + 1] + bb[c + 1]);
  hp[2] = (_Float16)(dz * rs * w[c + 2] + bb[c + 2]);
  hp[3] = (_Float16)(dw * rs * w[c + 3] + bb[c + 3]);
}

// ---------------------------------------------------------------------------
// 3) GEMM Win: (NTOK x 128) x (128 x 512) -> xc (first 256 cols), z (last 256)
// ---------------------------------------------------------------------------
__global__ void __launch_bounds__(128)
k_gemm_win(const _Float16* __restrict__ h, const _Float16* __restrict__ W,
           float* __restrict__ xc, float* __restrict__ z) {
  int wave = threadIdx.x >> 5, lane = threadIdx.x & 31;
  int m = lane & 15, half = lane >> 4;
  int mtile = blockIdx.x;
  const _Float16* arow = h + (size_t)mtile * 16 * DIMX;
  v16h a[4];
#pragma unroll
  for (int kc = 0; kc < 4; ++kc) a[kc] = load_a_frag(arow, DIMX, m, half, kc * 32);
  int nbase0 = blockIdx.y * 256 + wave * 64;
#pragma unroll
  for (int nt = 0; nt < 4; ++nt) {
    int nbase = nbase0 + nt * 16;
    v8f c = {};
#pragma unroll
    for (int kc = 0; kc < 4; ++kc) {
      v16h bf = load_b_frag(W, DIMX, nbase + m, half, kc * 32);
      c = wmma_f16(a[kc], bf, c);
    }
    int col = nbase + m;
    float* dst = (col < DI) ? xc : z;
    int c2 = col & (DI - 1);
#pragma unroll
    for (int r = 0; r < 8; ++r) {
      int row = mtile * 16 + r + 8 * half;
      dst[(size_t)row * DI + c2] = c[r];
    }
  }
}

// ---------------------------------------------------------------------------
// 4) depthwise causal conv (K=4) + bias + SiLU -> f16
// ---------------------------------------------------------------------------
__global__ void k_conv(const float* __restrict__ xc, const float* __restrict__ cw,
                       const float* __restrict__ cb, _Float16* __restrict__ xc16) {
  int idx = blockIdx.x * blockDim.x + threadIdx.x;   // NTOK*DI
  int tok = idx >> 8, d = idx & 255;
  int t = tok & 2047;
  float acc = cb[d];
#pragma unroll
  for (int k = 0; k < 4; ++k) {
    int back = 3 - k;
    if (t >= back) acc += xc[(size_t)(tok - back) * DI + d] * cw[d * 4 + k];
  }
  float sg = 1.f / (1.f + __expf(-acc));
  xc16[idx] = (_Float16)(acc * sg);
}

// ---------------------------------------------------------------------------
// 5) GEMM Wx: (NTOK x 256) x (256 x 48) -> xdbl
// ---------------------------------------------------------------------------
__global__ void __launch_bounds__(96)
k_gemm_wx(const _Float16* __restrict__ xc16, const _Float16* __restrict__ W,
          float* __restrict__ xdbl) {
  int wave = threadIdx.x >> 5, lane = threadIdx.x & 31;
  int m = lane & 15, half = lane >> 4;
  int mtile = blockIdx.x;
  const _Float16* arow = xc16 + (size_t)mtile * 16 * DI;
  int nbase = wave * 16;
  v8f c = {};
#pragma unroll
  for (int kc = 0; kc < 8; ++kc) {
    v16h a  = load_a_frag(arow, DI, m, half, kc * 32);
    v16h bf = load_b_frag(W, DI, nbase + m, half, kc * 32);
    c = wmma_f16(a, bf, c);
  }
  int col = nbase + m;
#pragma unroll
  for (int r = 0; r < 8; ++r) {
    int row = mtile * 16 + r + 8 * half;
    xdbl[(size_t)row * NXP + col] = c[r];
  }
}

// ---------------------------------------------------------------------------
// 6) selective scan with TDM-staged xdbl tiles.
//    One lane per (b, d, s); 16-lane shfl_xor reduction for y = sum_s h*C.
//    Wave 0 of each block issues tensor_load_to_lds (2D tile: 64 rows x 48 f32,
//    stride 48) into a double buffer; DMA of chunk c+1 overlaps compute of c.
//    D# encoding per CDNA5 ISA 8.3/8.4: group0 {count, lds_addr, global_addr,
//    type=2}, group1 {data_size=4B, tensor_dim0=48, tensor_dim1=65536,
//    tile_dim0=48, tile_dim1=64, tensor_dim0_stride=48}.
// ---------------------------------------------------------------------------
__global__ void __launch_bounds__(256)
k_scan(const float* __restrict__ xdbl, const _Float16* __restrict__ xc16,
       const float* __restrict__ z, const float* __restrict__ Wdt,
       const float* __restrict__ bdt, const float* __restrict__ A_log,
       const float* __restrict__ Dp, _Float16* __restrict__ y16) {
  __shared__ __align__(16) float sbuf[2][TCH * NXP];

  int b = blockIdx.x >> 4;
  int d = ((blockIdx.x & 15) << 4) + (threadIdx.x >> 4);
  int s = threadIdx.x & 15;
  float A = -__expf(A_log[d * DS + s]);
  float wr[RK];
#pragma unroll
  for (int r = 0; r < RK; ++r) wr[r] = Wdt[d * RK + r];
  float bd = bdt[d], Dd = Dp[d];
  size_t tokbase = ((size_t)b) << 11;

  unsigned lds0 = (unsigned)(unsigned long long)(&sbuf[0][0]);
  unsigned lds1 = (unsigned)(unsigned long long)(&sbuf[1][0]);
  unsigned long long gbase =
      (unsigned long long)(const void*)(xdbl + tokbase * NXP);

  u32x8 g1;
  g1[0] = 0x20000u;                 // data_size = 4 bytes; no multicast/iterate/pad
  g1[1] = (48u << 16);              // tensor_dim0 = 48 (elements)
  g1[2] = 0u;                       // tensor_dim1[15:0] = 0
  g1[3] = 1u | (48u << 16);         // tensor_dim1 = 65536 (no OOB), tile_dim0 = 48
  g1[4] = (unsigned)TCH;            // tile_dim1 = 64 rows
  g1[5] = 48u;                      // tensor_dim0_stride = 48 elements
  g1[6] = 0u;
  g1[7] = 0u;

  const bool issuer = (threadIdx.x < 32);   // wave 0 issues TDM (uniform branch)
  auto issue = [&](int chunk, unsigned ldsaddr) {
    unsigned long long ga =
        gbase + (unsigned long long)chunk * (TCH * NXP * sizeof(float));
    u32x4 g0;
    g0[0] = 1u;                                          // count = 1 (valid D#)
    g0[1] = ldsaddr;                                     // LDS byte address
    g0[2] = (unsigned)(ga & 0xffffffffu);                // global_addr[31:0]
    g0[3] = (unsigned)((ga >> 32) & 0x1ffffffu) | (2u << 30);  // addr hi | type=2
    asm volatile("tensor_load_to_lds %0, %1" :: "s"(g0), "s"(g1) : "memory");
  };

  if (issuer) {
    issue(0, lds0);
    __builtin_amdgcn_s_wait_tensorcnt(0);
  }
  __syncthreads();

  float h = 0.f;
  const int nchunk = LTOT / TCH;
  for (int c = 0; c < nchunk; ++c) {
    if (issuer && (c + 1) < nchunk) issue(c + 1, (c & 1) ? lds0 : lds1);
    const float* buf = sbuf[c & 1];
    int tb = c * TCH;
    for (int tt = 0; tt < TCH; ++tt) {
      int t = tb + tt;
      const float* row = buf + tt * NXP;
      float4 x0 = *(const float4*)(row);         // ds_load_b128
      float4 x1 = *(const float4*)(row + 4);
      float dtl = bd + wr[0] * x0.x + wr[1] * x0.y + wr[2] * x0.z + wr[3] * x0.w
                     + wr[4] * x1.x + wr[5] * x1.y + wr[6] * x1.z + wr[7] * x1.w;
      float dt = fmaxf(dtl, 0.f) + __logf(1.f + __expf(-fabsf(dtl)));  // softplus
      float u  = (float)xc16[(tokbase + t) * DI + d];
      float Bv = row[8 + s];
      float Cv = row[24 + s];
      float dA = __expf(dt * A);
      h = dA * h + (dt * u) * Bv;
      float p = h * Cv;
#pragma unroll
      for (int o = 8; o > 0; o >>= 1) p += __shfl_xor(p, o, 16);
      if (s == 0) {
        float zv = z[(tokbase + t) * DI + d];
        float y  = p + u * Dd;
        float g  = zv / (1.f + __expf(-zv));     // silu(z)
        y16[(tokbase + t) * DI + d] = (_Float16)(y * g);
      }
    }
    if (issuer) __builtin_amdgcn_s_wait_tensorcnt(0);
    __syncthreads();
  }
}

// ---------------------------------------------------------------------------
// 7) GEMM Wout: (NTOK x 256) x (256 x 128) + residual, un-reverse, accumulate
// ---------------------------------------------------------------------------
__global__ void __launch_bounds__(128)
k_gemm_wout(const _Float16* __restrict__ y16, const _Float16* __restrict__ W,
            const float* __restrict__ xin, float* __restrict__ xout,
            int rev, int accum) {
  int wave = threadIdx.x >> 5, lane = threadIdx.x & 31;
  int m = lane & 15, half = lane >> 4;
  int mtile = blockIdx.x;
  const _Float16* arow = y16 + (size_t)mtile * 16 * DI;
  v16h a[8];
#pragma unroll
  for (int kc = 0; kc < 8; ++kc) a[kc] = load_a_frag(arow, DI, m, half, kc * 32);
#pragma unroll
  for (int nt = 0; nt < 2; ++nt) {
    int nbase = wave * 32 + nt * 16;
    v8f c = {};
#pragma unroll
    for (int kc = 0; kc < 8; ++kc) {
      v16h bf = load_b_frag(W, DI, nbase + m, half, kc * 32);
      c = wmma_f16(a[kc], bf, c);
    }
    int col = nbase + m;
#pragma unroll
    for (int r = 0; r < 8; ++r) {
      int rowtok = mtile * 16 + r + 8 * half;
      int bb = rowtok >> 11, tt = rowtok & 2047;
      int ot = rev ? (2047 - tt) : tt;
      size_t idx = (((size_t)(bb << 11)) + ot) * DIMX + col;
      float val = xin[idx] + c[r];
      if (accum) xout[idx] += val;
      else       xout[idx] = val;
    }
  }
}

// ---------------------------------------------------------------------------
// host driver
// ---------------------------------------------------------------------------
extern "C" void kernel_launch(void* const* d_in, const int* in_sizes, int n_in,
                              void* d_out, int out_size, void* d_ws, size_t ws_size,
                              hipStream_t stream) {
  (void)in_sizes; (void)n_in; (void)out_size; (void)ws_size;
  const float* x_adap = (const float*)d_in[0];
  const float* xi_adap = (const float*)d_in[1];
  const float* ln_w  = (const float*)d_in[2];
  const float* ln_b  = (const float*)d_in[3];
  const float* Win   = (const float*)d_in[4];
  const float* convw = (const float*)d_in[5];
  const float* convb = (const float*)d_in[6];
  const float* Wx    = (const float*)d_in[7];
  const float* Wdt   = (const float*)d_in[8];
  const float* bdt   = (const float*)d_in[9];
  const float* A_log = (const float*)d_in[10];
  const float* Dp    = (const float*)d_in[11];
  const float* Wout  = (const float*)d_in[12];

  char* ws = (char*)d_ws;
  size_t off = 0;
  auto alloc = [&](size_t bytes) {
    char* p = ws + off;
    off = (off + bytes + 255) & ~(size_t)255;
    return (void*)p;
  };
  float*     x0     = (float*)alloc((size_t)NTOK * DIMX * 4);
  float*     x1     = (float*)alloc((size_t)NTOK * DIMX * 4);
  _Float16*  h16    = (_Float16*)alloc((size_t)NTOK * DIMX * 2);
  float*     xc     = (float*)alloc((size_t)NTOK * DI * 4);
  float*     zb     = (float*)alloc((size_t)NTOK * DI * 4);
  _Float16*  xc16   = (_Float16*)alloc((size_t)NTOK * DI * 2);
  float*     xdbl   = (float*)alloc((size_t)NTOK * NXP * 4);
  _Float16*  y16    = (_Float16*)alloc((size_t)NTOK * DI * 2);
  _Float16*  Win16  = (_Float16*)alloc((size_t)512 * DIMX * 2);
  _Float16*  Wx16   = (_Float16*)alloc((size_t)NXP * DI * 2);
  _Float16*  Wout16 = (_Float16*)alloc((size_t)DIMX * DI * 2);

  k_concat<<<(NTOK * DIMX) / 256, 256, 0, stream>>>(x_adap, xi_adap, x0);

  float* cur = x0;
  const int wtot = 512 * DIMX + NXP * DI + DIMX * DI;   // 110592
  for (int i = 0; i < 2; ++i) {
    float* acc = (i == 0) ? x1 : (float*)d_out;
    k_wconv<<<(wtot + 255) / 256, 256, 0, stream>>>(
        Win + (size_t)i * 512 * DIMX, Wx + (size_t)i * (RK + 2 * DS) * DI,
        Wout + (size_t)i * DIMX * DI, Win16, Wx16, Wout16);
    for (int dir = 0; dir < 2; ++dir) {
      k_ln<<<NTOK / 8, 256, 0, stream>>>(cur, ln_w + i * DIMX, ln_b + i * DIMX,
                                         h16, dir);
      k_gemm_win<<<dim3(NTOK / 16, 2), 128, 0, stream>>>(h16, Win16, xc, zb);
      k_conv<<<(NTOK * DI) / 256, 256, 0, stream>>>(xc, convw + i * DI * 4,
                                                    convb + i * DI, xc16);
      k_gemm_wx<<<NTOK / 16, 96, 0, stream>>>(xc16, Wx16, xdbl);
      k_scan<<<BSZ * (DI / 16), 256, 0, stream>>>(
          xdbl, xc16, zb, Wdt + i * DI * RK, bdt + i * DI,
          A_log + i * DI * DS, Dp + i * DI, y16);
      k_gemm_wout<<<NTOK / 16, 128, 0, stream>>>(y16, Wout16, cur, acc, dir, dir);
    }
    cur = acc;
  }
}
